// TemporalAttention_49349174231050
// MI455X (gfx1250) — matrix operand order repaired and checked by reading, not compiled, not run on hardware
//
#include <hip/hip_runtime.h>
#include <hip/hip_bf16.h>

// ---------------------------------------------------------------------------
// Temporal attention:  l[b,t] = v_d . tanh(dsp[b] + encoder_h[b,t] @ U_d)
//                      out    = softmax_T(l)
// B=64, T=2048, M=P=512.  Heavy op: (B*T,512)x(512,512) GEMM in bf16 WMMA
// with fused hw-tanh/v_d reduction epilogue (never materializes 256MB tensor).
// ---------------------------------------------------------------------------

#define B_DIM 64
#define T_DIM 2048
#define M_DIM 512
#define ROWS_PER_WG 64          // 4 waves x 16-row strip each
#define GEMM_THREADS 128

typedef __attribute__((ext_vector_type(16))) __bf16 v16bf;
typedef __attribute__((ext_vector_type(8)))  float  v8f;

struct U8x32 { uint4 lo, hi; };   // 32 bytes == v16bf

__device__ __forceinline__ unsigned short f2bf(float x) {
    unsigned int u = __builtin_bit_cast(unsigned int, x);
    unsigned int r = u + 0x7FFFu + ((u >> 16) & 1u);   // round-to-nearest-even
    return (unsigned short)(r >> 16);
}

// Hardware tanh (CDNA5 V_TANH_F32 transcendental) instead of OCML polynomial.
__device__ __forceinline__ float hw_tanh(float x) {
#if __has_builtin(__builtin_amdgcn_tanhf)
    return __builtin_amdgcn_tanhf(x);
#else
    float y;
    // v_nop covers the TRANS-op result hazard (1 op gap before use).
    asm volatile("v_tanh_f32 %0, %1\n\tv_nop" : "=v"(y) : "v"(x));
    return y;
#endif
}

// ---------------------------------------------------------------------------
// Kernel 1: dsp[b,m] = sum_k concat(h,c)[b,k] * W_d[k,m] + b_wd[m] + b_ud[m]
// ---------------------------------------------------------------------------
__global__ void __launch_bounds__(256)
dsp_kernel(const float* __restrict__ h, const float* __restrict__ c,
           const float* __restrict__ Wd, const float* __restrict__ bwd,
           const float* __restrict__ bud, float* __restrict__ dsp)
{
    const int m = blockIdx.x * 256 + threadIdx.x;   // 0..511
    const int b = blockIdx.y;
    float s = 0.f;
    for (int k = 0; k < M_DIM; ++k)
        s += h[b * M_DIM + k] * Wd[(size_t)k * M_DIM + m];
    for (int k = 0; k < M_DIM; ++k)
        s += c[b * M_DIM + k] * Wd[(size_t)(M_DIM + k) * M_DIM + m];
    dsp[b * M_DIM + m] = s + bwd[m] + bud[m];
}

// ---------------------------------------------------------------------------
// Kernel 2: fused GEMM (bf16 WMMA) + tanh + v_d reduction -> l[b*T+t]
// LDS layout (dynamic, 135168 B):
//   [0      .. 65535 ]  A strip : 64 rows x 512 K, bf16
//   [65536  .. 131071]  B panel : 64 n x 512 k (n-major), bf16  (per n-group)
//   [131072 .. 133119]  dsp row : 512 f32
//   [133120 .. 135167]  v_d     : 512 f32
// Barrier-free inner K loop: whole-K B panel staged once per n-group.
// ---------------------------------------------------------------------------
extern __shared__ uint4 smem_u4[];

__global__ void __launch_bounds__(GEMM_THREADS)
attn_gemm_kernel(const float* __restrict__ enc,   // (B*T, M)
                 const float* __restrict__ Ud,    // (M, M)
                 const float* __restrict__ dsp,   // (B, M)
                 const float* __restrict__ vd,    // (M,)
                 float* __restrict__ out_l)       // (B*T,)
{
    char* smem = (char*)smem_u4;
    unsigned short* ldsA = (unsigned short*)smem;              // 64x512 bf16
    unsigned short* ldsB = (unsigned short*)(smem + 65536);    // [n(64)][k(512)] bf16
    float* ldsD = (float*)(smem + 131072);                     // dsp row
    float* ldsV = (float*)(smem + 133120);                     // v_d

    const int tid     = threadIdx.x;
    const int wg_row0 = blockIdx.x * ROWS_PER_WG;
    const int b       = wg_row0 / T_DIM;   // 64 | 2048 -> uniform per WG

    // ---- stage A strip (f32 -> bf16), float4 coalesced loads ----
    {
        const float4* src = (const float4*)(enc + (size_t)wg_row0 * M_DIM);
        #pragma unroll 4
        for (int i = 0; i < (ROWS_PER_WG * M_DIM / 4) / GEMM_THREADS; ++i) {
            int e4 = i * GEMM_THREADS + tid;
            float4 v = src[e4];
            unsigned int p0 = (unsigned int)f2bf(v.x) | ((unsigned int)f2bf(v.y) << 16);
            unsigned int p1 = (unsigned int)f2bf(v.z) | ((unsigned int)f2bf(v.w) << 16);
            *(uint2*)(ldsA + e4 * 4) = make_uint2(p0, p1);
        }
    }
    // ---- stage dsp row + v_d ----
    for (int i = 0; i < M_DIM / GEMM_THREADS; ++i) {
        int idx = i * GEMM_THREADS + tid;
        ldsD[idx] = dsp[b * M_DIM + idx];
        ldsV[idx] = vd[idx];
    }

    const int wave  = tid >> 5;
    const int lane  = tid & 31;
    const int hl    = lane >> 4;      // half-wave select
    const int lrow  = lane & 15;
    const int strip = wave * 16;      // local row base of this wave's strip

    float acc[8];
    #pragma unroll
    for (int v = 0; v < 8; ++v) acc[v] = 0.f;

    const v8f vzero = {0.f, 0.f, 0.f, 0.f, 0.f, 0.f, 0.f, 0.f};

    for (int g = 0; g < M_DIM / 64; ++g) {          // 8 n-groups of 64 cols
        const int gbase = g * 64;

        // ---- stage full-K B panel for this n-group: ldsB[n][k] ----
        __syncthreads();                // previous group's compute done
        #pragma unroll 4
        for (int i = 0; i < (M_DIM * 64) / GEMM_THREADS; ++i) {
            int e = i * GEMM_THREADS + tid;
            int k = e >> 6;             // 0..511
            int n = e & 63;
            ldsB[n * M_DIM + k] = f2bf(Ud[(size_t)k * M_DIM + gbase + n]);
        }
        __syncthreads();

        v8f C[4];
        #pragma unroll
        for (int t = 0; t < 4; ++t) C[t] = vzero;

        // ---- barrier-free K loop ----
        for (int kc = 0; kc < M_DIM / 32; ++kc) {   // 16 K-chunks of 32
            const int k0 = kc * 32;

            // A fragment: row = strip+lrow; lanes<16 hold K {0-7,16-23},
            // lanes>=16 hold K {8-15,24-31} (ISA 16-bit A 16x32 layout)
            const unsigned short* arow = ldsA + (strip + lrow) * M_DIM;
            const int ka = k0 + (hl ? 8 : 0);
            uint4 a_lo = *(const uint4*)(arow + ka);
            uint4 a_hi = *(const uint4*)(arow + ka + 16);
            v16bf afrag = __builtin_bit_cast(v16bf, U8x32{a_lo, a_hi});

            // Load ALL 4 B fragments first (distinct regs) so the scheduler
            // can batch the ds_loads and issue the 4 WMMAs back-to-back.
            v16bf bfrag[4];
            #pragma unroll
            for (int t = 0; t < 4; ++t) {
                // B fragment: col = t*16+lrow; lanes<16 K {0-7,16-23}, lanes>=16 K {8-15,24-31}
                const unsigned short* bcol =
                    ldsB + (t * 16 + lrow) * M_DIM + k0 + hl * 8;
                uint4 b_lo = *(const uint4*)(bcol);
                uint4 b_hi = *(const uint4*)(bcol + 16);
                bfrag[t] = __builtin_bit_cast(v16bf, U8x32{b_lo, b_hi});
            }
            #pragma unroll
            for (int t = 0; t < 4; ++t)
                C[t] = __builtin_amdgcn_wmma_f32_16x16x32_bf16(
                           false, afrag, false, bfrag[t], (short)0, C[t],
                           false, false);
        }

        // fused epilogue: acc[row] += tanh(C + dsp[n]) * v_d[n] (hardware tanh)
        #pragma unroll
        for (int t = 0; t < 4; ++t) {
            const int n = gbase + t * 16 + lrow;
            const float d = ldsD[n];
            const float w = ldsV[n];
            #pragma unroll
            for (int v = 0; v < 8; ++v)
                acc[v] += hw_tanh(C[t][v] + d) * w;
        }
    }

    // cross-lane reduce (16 cols per half-wave), write l per row
    #pragma unroll
    for (int v = 0; v < 8; ++v) {
        float s = acc[v];
        s += __shfl_xor(s, 1);
        s += __shfl_xor(s, 2);
        s += __shfl_xor(s, 4);
        s += __shfl_xor(s, 8);
        if (lrow == 0) {
            const int row = wg_row0 + strip + v + 8 * hl;  // C layout: M = v + 8*half
            out_l[row] = s;
        }
    }
}

// ---------------------------------------------------------------------------
// Kernel 3: softmax over T per batch (b_vd dropped: softmax shift-invariant)
// ---------------------------------------------------------------------------
__global__ void __launch_bounds__(256)
softmax_kernel(const float* __restrict__ l, float* __restrict__ out)
{
    __shared__ float red[256];
    const int b = blockIdx.x, tid = threadIdx.x;
    const float* lb = l + b * T_DIM;

    float m = -3.4e38f;
    for (int t = tid; t < T_DIM; t += 256) m = fmaxf(m, lb[t]);
    red[tid] = m; __syncthreads();
    for (int s = 128; s > 0; s >>= 1) {
        if (tid < s) red[tid] = fmaxf(red[tid], red[tid + s]);
        __syncthreads();
    }
    const float gmax = red[0]; __syncthreads();

    float sum = 0.f;
    for (int t = tid; t < T_DIM; t += 256) sum += expf(lb[t] - gmax);
    red[tid] = sum; __syncthreads();
    for (int s = 128; s > 0; s >>= 1) {
        if (tid < s) red[tid] += red[tid + s];
        __syncthreads();
    }
    const float inv = 1.f / red[0];

    for (int t = tid; t < T_DIM; t += 256)
        out[b * T_DIM + t] = expf(lb[t] - gmax) * inv;
}

// ---------------------------------------------------------------------------
extern "C" void kernel_launch(void* const* d_in, const int* in_sizes, int n_in,
                              void* d_out, int out_size, void* d_ws, size_t ws_size,
                              hipStream_t stream)
{
    const float* h   = (const float*)d_in[0];  // hidden_state (B,P)
    const float* c   = (const float*)d_in[1];  // cell_state   (B,P)
    const float* enc = (const float*)d_in[2];  // encoder_h    (B,T,M)
    const float* Wd  = (const float*)d_in[3];  // W_d (2P,M)
    const float* bwd = (const float*)d_in[4];  // b_wd (M,)
    const float* Ud  = (const float*)d_in[5];  // U_d (M,M)
    const float* bud = (const float*)d_in[6];  // b_ud (M,)
    const float* vd  = (const float*)d_in[7];  // v_d (M,1)
    // d_in[8] = b_vd: scalar shift before softmax -> no effect, ignored.

    float* dsp  = (float*)d_ws;                                    // 64*512 f32
    float* lbuf = (float*)((char*)d_ws + B_DIM * M_DIM * 4);       // 131072 f32
    float* out  = (float*)d_out;

    dsp_kernel<<<dim3(2, B_DIM), 256, 0, stream>>>(h, c, Wd, bwd, bud, dsp);

    const int n_wg = (B_DIM * T_DIM) / ROWS_PER_WG;   // 2048
    attn_gemm_kernel<<<dim3(n_wg), GEMM_THREADS, 135168, stream>>>(enc, Ud, dsp, vd, lbuf);

    softmax_kernel<<<dim3(B_DIM), 256, 0, stream>>>(lbuf, out);
}
